// Enformer_48713519071261
// MI455X (gfx1250) — matrix operand order repaired and checked by reading, not compiled, not run on hardware
//
#include <hip/hip_runtime.h>
#include <hip/hip_bf16.h>
#include <math.h>

// ---------------------------------------------------------------------------
// Enformer forward for MI455X (gfx1250, wave32, WMMA).
// All GEMMs run through a single templated WMMA kernel:
//   V_WMMA_F32_16X16X32_F16, f32 accumulate, f32->f16 convert on LDS stage.
// LDS layouts chosen so every fragment load is ds_load_b128:
//   As[BM][BK+8]  row-major   (A fragment: two contiguous 8-half runs)
//   Bs[BN][BK+8]  transposed  (B fragment: two contiguous 8-half runs)
// A-tile staging uses GLOBAL_LOAD_ASYNC_TO_LDS_B128 + s_wait_asynccnt.
// ---------------------------------------------------------------------------

typedef __attribute__((ext_vector_type(16))) _Float16 v16h;
typedef __attribute__((ext_vector_type(8)))  _Float16 v8h;
typedef __attribute__((ext_vector_type(4)))  _Float16 v4h;
typedef __attribute__((ext_vector_type(8)))  float    v8f;

#define BM 64
#define BN 128
#define BK 32
#define PADH 8   // LDS row padding in halves (keeps 16B alignment, spreads banks)

#define ACT_NONE     0
#define ACT_RELU     1
#define ACT_GELU     2
#define ACT_SOFTPLUS 3

#if defined(__gfx1250__) && \
    __has_builtin(__builtin_amdgcn_global_load_async_to_lds_b128) && \
    __has_builtin(__builtin_amdgcn_s_wait_asynccnt)
#define USE_ASYNC_STAGE 1
typedef int v4i_t __attribute__((ext_vector_type(4)));
typedef __attribute__((address_space(1))) v4i_t as1_v4i;
typedef __attribute__((address_space(3))) v4i_t as3_v4i;
#else
#define USE_ASYNC_STAGE 0
#endif

__device__ __forceinline__ float act_apply(float v, int act) {
  if (act == ACT_RELU)     return v > 0.f ? v : 0.f;
  if (act == ACT_GELU)     return v / (1.f + __expf(-1.702f * v));
  if (act == ACT_SOFTPLUS) return (v > 20.f) ? v : log1pf(__expf(v));
  return v;
}

// C[M,N] = act( alpha*(A op B) + bias + resid (+ C if accumulate) )
// TA: logical A[m][k] = Aptr[k*lda + m]   (else Aptr[m*lda + k])
// TB: logical B[k][n] = Bptr[n*ldb + k]   (else Bptr[k*ldb + n])
// Requirements guaranteed by the caller: K % 32 == 0; contiguous-side base
// pointers and leading dims are 16B aligned (all lda/ldb are multiples of 4).
template <bool TA, bool TB>
__global__ __launch_bounds__(256)
void gemm_wmma_kernel(const float* __restrict__ A, const float* __restrict__ B,
                      float* __restrict__ C, const float* __restrict__ bias,
                      const float* __restrict__ resid,
                      int M, int N, int K, int lda, int ldb, int ldc,
                      int bias_mode, int accumulate, int act, float alpha) {
  __shared__ _Float16 As[BM][BK + PADH];
  __shared__ _Float16 Bs[BN][BK + PADH];
#if USE_ASYNC_STAGE
  __shared__ float Asf[BM][BK];
#endif

  const int tid  = threadIdx.x;
  const int lane = tid & 31;
  const int wave = tid >> 5;
  const int wmb  = (wave >> 2) * 32;   // 2 waves along M
  const int wnb  = (wave & 3) * 32;    // 4 waves along N
  const int m0   = blockIdx.y * BM;
  const int n0   = blockIdx.x * BN;
  const int r    = lane & 15;
  const int hf   = lane >> 4;

  v8f acc[2][2] = {};

  for (int k0 = 0; k0 < K; k0 += BK) {
    if (tid == 0 && k0 + BK < K) {
      __builtin_prefetch(TA ? (const void*)&A[(size_t)(k0 + BK) * lda + m0]
                            : (const void*)&A[(size_t)m0 * lda + (k0 + BK)], 0, 1);
      __builtin_prefetch(TB ? (const void*)&B[(size_t)n0 * ldb + (k0 + BK)]
                            : (const void*)&B[(size_t)(k0 + BK) * ldb + n0], 0, 1);
    }

    // ------------------ stage A tile (BM x BK) -> f16 ------------------
    if (TA) {
      // A stored [K][M]; read coalesced along m, scatter-transpose into As.
      for (int i = tid; i < BK * (BM / 4); i += 256) {
        int k  = i >> 4;          // BM/4 == 16
        int mq = (i & 15) * 4;
        int gm = m0 + mq;
        float4 v = {0.f, 0.f, 0.f, 0.f};
        if (gm + 3 < M) v = *(const float4*)&A[(size_t)(k0 + k) * lda + gm];
        As[mq + 0][k] = (_Float16)v.x;
        As[mq + 1][k] = (_Float16)v.y;
        As[mq + 2][k] = (_Float16)v.z;
        As[mq + 3][k] = (_Float16)v.w;
      }
    } else {
#if USE_ASYNC_STAGE
      // async-copy raw f32 rows into LDS (16B per lane), then convert.
      for (int i = tid; i < BM * (BK / 4); i += 256) {
        int m = i >> 3, kq = (i & 7) * 4;
        int gm = m0 + m;
        const float* gp = &A[(size_t)(gm < M ? gm : 0) * lda + k0 + kq];
        __builtin_amdgcn_global_load_async_to_lds_b128(
            (as1_v4i*)gp, (as3_v4i*)&Asf[m][kq], 0, 0);
      }
      __builtin_amdgcn_s_wait_asynccnt(0);
      for (int i = tid; i < BM * (BK / 4); i += 256) {
        int m = i >> 3, kq = (i & 7) * 4;
        float4 v = {0.f, 0.f, 0.f, 0.f};
        if (m0 + m < M) v = *(const float4*)&Asf[m][kq];
        v4h h; h[0] = (_Float16)v.x; h[1] = (_Float16)v.y;
               h[2] = (_Float16)v.z; h[3] = (_Float16)v.w;
        *(v4h*)&As[m][kq] = h;
      }
#else
      for (int i = tid; i < BM * (BK / 4); i += 256) {
        int m = i >> 3, kq = (i & 7) * 4;
        int gm = m0 + m;
        float4 v = {0.f, 0.f, 0.f, 0.f};
        if (gm < M) v = *(const float4*)&A[(size_t)gm * lda + k0 + kq];
        v4h h; h[0] = (_Float16)v.x; h[1] = (_Float16)v.y;
               h[2] = (_Float16)v.z; h[3] = (_Float16)v.w;
        *(v4h*)&As[m][kq] = h;
      }
#endif
    }

    // --------------- stage B tile (BK x BN) -> f16, transposed ---------------
    if (TB) {
      // B stored [N][K]; contiguous along k -> vector load + vector LDS store.
      for (int i = tid; i < BN * (BK / 4); i += 256) {
        int n = i >> 3, kq = (i & 7) * 4;
        int gn = n0 + n;
        float4 v = {0.f, 0.f, 0.f, 0.f};
        if (gn < N) v = *(const float4*)&B[(size_t)gn * ldb + k0 + kq];
        v4h h; h[0] = (_Float16)v.x; h[1] = (_Float16)v.y;
               h[2] = (_Float16)v.z; h[3] = (_Float16)v.w;
        *(v4h*)&Bs[n][kq] = h;
      }
    } else {
      // B stored [K][N]; coalesced f32 reads along n, scatter-transpose.
      for (int i = tid; i < (BK * BN) / 4; i += 256) {
        int k  = i >> 5;          // BN/4 == 32
        int nq = (i & 31) * 4;
        const float* bp = &B[(size_t)(k0 + k) * ldb + n0 + nq];
#pragma unroll
        for (int j = 0; j < 4; ++j) {
          int gn = n0 + nq + j;
          Bs[nq + j][k] = (_Float16)((gn < N) ? bp[j] : 0.f);
        }
      }
    }
    __syncthreads();

    // --------------- fragments (all ds_load_b128) + WMMA ---------------
    v16h af[2], bf[2];
#pragma unroll
    for (int mt = 0; mt < 2; ++mt) {
      const _Float16* ap = &As[wmb + mt * 16 + r][hf * 8];
      v8h lo = *(const v8h*)ap;
      v8h hi = *(const v8h*)(ap + 16);
      af[mt] = __builtin_shufflevector(lo, hi, 0, 1, 2, 3, 4, 5, 6, 7,
                                       8, 9, 10, 11, 12, 13, 14, 15);
    }
#pragma unroll
    for (int nt = 0; nt < 2; ++nt) {
      const _Float16* bp = &Bs[wnb + nt * 16 + r][hf * 8];
      v8h lo = *(const v8h*)bp;
      v8h hi = *(const v8h*)(bp + 16);
      bf[nt] = __builtin_shufflevector(lo, hi, 0, 1, 2, 3, 4, 5, 6, 7,
                                       8, 9, 10, 11, 12, 13, 14, 15);
    }
#pragma unroll
    for (int mt = 0; mt < 2; ++mt)
#pragma unroll
      for (int nt = 0; nt < 2; ++nt)
        acc[mt][nt] = __builtin_amdgcn_wmma_f32_16x16x32_f16(
            false, af[mt], false, bf[nt], (short)0, acc[mt][nt], false, false);
    __syncthreads();
  }

  // epilogue: C/D layout -> lane = N, VGPR index = M (+8 for high half-wave)
#pragma unroll
  for (int mt = 0; mt < 2; ++mt)
#pragma unroll
    for (int nt = 0; nt < 2; ++nt)
#pragma unroll
      for (int rr = 0; rr < 8; ++rr) {
        int gm = m0 + wmb + mt * 16 + hf * 8 + rr;
        int gn = n0 + wnb + nt * 16 + r;
        if (gm < M && gn < N) {
          size_t idx = (size_t)gm * ldc + gn;
          float v = alpha * acc[mt][nt][rr];
          if (bias_mode == 1)      v += bias[gn];
          else if (bias_mode == 2) v += bias[gm];
          if (resid)      v += resid[idx];
          if (accumulate) v += C[idx];
          C[idx] = act_apply(v, act);
        }
      }
}

// ---------------------------------------------------------------------------
// Elementwise / reduction kernels
// ---------------------------------------------------------------------------

__global__ void stem_conv_kernel(const int* __restrict__ ids, const float* __restrict__ w,
                                 const float* __restrict__ b, float* __restrict__ out, int L) {
  size_t idx = (size_t)blockIdx.x * blockDim.x + threadIdx.x;
  if (idx >= (size_t)768 * L) return;
  int co = (int)(idx / L);
  int n  = (int)(idx % L);
  float acc = b[co];
#pragma unroll
  for (int t = 0; t < 15; ++t) {
    int p = n + t - 7;
    if (p >= 0 && p < L) {
      int cls = ids[p];
      acc += w[((size_t)co * 4 + cls) * 15 + t];
    }
  }
  out[idx] = acc;
}

// y[c][n+pad] = gelu(batchnorm(x[c][n])); zero padding at both ends
__global__ void bn_gelu_pad_kernel(const float* __restrict__ x, const float* __restrict__ g,
                                   const float* __restrict__ b, const float* __restrict__ rm,
                                   const float* __restrict__ rv, float* __restrict__ y,
                                   int C, int L, int pad) {
  int Lp = L + 2 * pad;
  size_t idx = (size_t)blockIdx.x * blockDim.x + threadIdx.x;
  if (idx >= (size_t)C * Lp) return;
  int c = (int)(idx / Lp);
  int n = (int)(idx % Lp) - pad;
  float v = 0.f;
  if (n >= 0 && n < L) {
    float t = (x[(size_t)c * L + n] - rm[c]) * rsqrtf(rv[c] + 1e-5f) * g[c] + b[c];
    v = t / (1.f + __expf(-1.702f * t));
  }
  y[idx] = v;
}

// softmax over adjacent position pairs, weighted sum
__global__ void attn_pool_kernel(const float* __restrict__ x, const float* __restrict__ lg,
                                 float* __restrict__ y, int C, int L) {
  int Lh = L >> 1;
  size_t idx = (size_t)blockIdx.x * blockDim.x + threadIdx.x;
  if (idx >= (size_t)C * Lh) return;
  int c = (int)(idx / Lh);
  int m = (int)(idx % Lh);
  size_t b0 = (size_t)c * L + 2 * m;
  float l0 = lg[b0], l1 = lg[b0 + 1];
  float mx = fmaxf(l0, l1);
  float e0 = __expf(l0 - mx), e1 = __expf(l1 - mx);
  y[idx] = (x[b0] * e0 + x[b0 + 1] * e1) / (e0 + e1);
}

// repack conv5 weights (co,ci,5) -> wt[t][co][ci]
__global__ void repack_conv5_kernel(const float* __restrict__ w, float* __restrict__ wt,
                                    int co, int ci) {
  size_t idx = (size_t)blockIdx.x * blockDim.x + threadIdx.x;
  size_t tot = (size_t)co * ci * 5;
  if (idx >= tot) return;
  int t = (int)(idx % 5);
  size_t mk = idx / 5;
  wt[(size_t)t * co * ci + mk] = w[idx];
}

__global__ void transpose_kernel(const float* __restrict__ in, float* __restrict__ out,
                                 int R, int C) {
  size_t idx = (size_t)blockIdx.x * blockDim.x + threadIdx.x;
  if (idx >= (size_t)R * C) return;
  int r = (int)(idx / C), c = (int)(idx % C);
  out[(size_t)c * R + r] = in[idx];
}

__global__ __launch_bounds__(256)
void layernorm_kernel(const float* __restrict__ x, const float* __restrict__ g,
                      const float* __restrict__ b, float* __restrict__ y, int D) {
  int row = blockIdx.x;
  const float* xr = x + (size_t)row * D;
  float* yr = y + (size_t)row * D;
  __shared__ float red[256];
  int tid = threadIdx.x;
  float s = 0.f;
  for (int i = tid; i < D; i += 256) s += xr[i];
  red[tid] = s; __syncthreads();
  for (int o = 128; o > 0; o >>= 1) { if (tid < o) red[tid] += red[tid + o]; __syncthreads(); }
  float mu = red[0] / D;
  __syncthreads();
  float v = 0.f;
  for (int i = tid; i < D; i += 256) { float t = xr[i] - mu; v += t * t; }
  red[tid] = v; __syncthreads();
  for (int o = 128; o > 0; o >>= 1) { if (tid < o) red[tid] += red[tid + o]; __syncthreads(); }
  float rstd = rsqrtf(red[0] / D + 1e-5f);
  for (int i = tid; i < D; i += 256) yr[i] = (xr[i] - mu) * rstd * g[i] + b[i];
}

__global__ void add_bcast_row_kernel(const float* __restrict__ x, const float* __restrict__ bias,
                                     float* __restrict__ y, int rows, int cols) {
  size_t idx = (size_t)blockIdx.x * blockDim.x + threadIdx.x;
  if (idx >= (size_t)rows * cols) return;
  y[idx] = x[idx] + bias[idx % cols];
}

// logits[h,i,j] = content[h,i,j] + rel[h,i, 1023 + j - i]; softmax over j, in place
__global__ __launch_bounds__(256)
void attn_softmax_kernel(float* __restrict__ content, const float* __restrict__ rel) {
  int h = blockIdx.x >> 10;
  int i = blockIdx.x & 1023;
  float* crow = content + ((size_t)h * 1024 + i) * 1024;
  const float* rrow = rel + ((size_t)h * 1024 + i) * 2047;
  int tid = threadIdx.x;
  float vals[4];
  float mx = -1e30f;
#pragma unroll
  for (int t = 0; t < 4; ++t) {
    int j = tid + t * 256;
    float v = crow[j] + rrow[1023 + j - i];
    vals[t] = v; mx = fmaxf(mx, v);
  }
  __shared__ float red[256];
  red[tid] = mx; __syncthreads();
  for (int o = 128; o > 0; o >>= 1) { if (tid < o) red[tid] = fmaxf(red[tid], red[tid + o]); __syncthreads(); }
  mx = red[0]; __syncthreads();
  float s = 0.f;
#pragma unroll
  for (int t = 0; t < 4; ++t) { vals[t] = __expf(vals[t] - mx); s += vals[t]; }
  red[tid] = s; __syncthreads();
  for (int o = 128; o > 0; o >>= 1) { if (tid < o) red[tid] += red[tid + o]; __syncthreads(); }
  float inv = 1.f / red[0];
#pragma unroll
  for (int t = 0; t < 4; ++t) crow[tid + t * 256] = vals[t] * inv;
}

__global__ void init_max_kernel(unsigned int* p) {
  if (blockIdx.x == 0 && threadIdx.x == 0) *p = 0u;
}

// gamma basis un-normalized probs + global max (positive floats -> uint max ok)
__global__ void gamma_prob_kernel(float* __restrict__ prob, unsigned int* __restrict__ maxb) {
  int idx = blockIdx.x * blockDim.x + threadIdx.x;
  if (idx >= 2047 * 32) return;
  int p = idx / 32, j = idx % 32;
  float d = (float)p - 1023.f;
  float ad = fabsf(d);
  float mean = 32.f + (1024.f - 32.f) * (float)j / 31.f;
  float conc = (mean / 16.f) * (mean / 16.f);
  float rate = mean / 256.f;
  float pr = 0.f;
  if (ad > 0.f) {
    float lu = (conc - 1.f) * logf(ad) - rate * ad;
    float ln_ = lgammaf(conc) - conc * logf(rate);
    pr = __expf(lu - ln_);
  }
  pr += 1e-8f;
  prob[idx] = pr;
  atomicMax(maxb, __float_as_uint(pr));
}

// emb[p][f], f<192: [e_exp(32) | e_cm(32) | e_gamma(32) | sign(d)*same(96)]
__global__ void pos_emb_kernel(const float* __restrict__ prob, const unsigned int* __restrict__ maxb,
                               float* __restrict__ emb) {
  int idx = blockIdx.x * blockDim.x + threadIdx.x;
  if (idx >= 2047 * 192) return;
  int p = idx / 192, f = idx % 192;
  float d = (float)p - 1023.f;
  float ad = fabsf(d);
  float sgn = 1.f;
  int ff = f;
  if (f >= 96) { ff = f - 96; sgn = (d > 0.f) ? 1.f : ((d < 0.f) ? -1.f : 0.f); }
  int j = ff % 32, kind = ff / 32;
  float v;
  if (kind == 0) {
    float hl = exp2f(3.f + 7.f * (float)j / 31.f);
    v = exp2f(-ad / hl);
  } else if (kind == 1) {
    float wdt = exp2f((float)(j + 1)) - 1.f;
    v = (wdt > ad) ? 1.f : 0.f;
  } else {
    v = prob[p * 32 + j] / __uint_as_float(*maxb);
  }
  emb[idx] = sgn * v;
}

__global__ void gelu_kernel(float* __restrict__ x, size_t n) {
  size_t idx = (size_t)blockIdx.x * blockDim.x + threadIdx.x;
  if (idx >= n) return;
  float v = x[idx];
  x[idx] = v / (1.f + __expf(-1.702f * v));
}

// ---------------------------------------------------------------------------
// Host orchestration
// ---------------------------------------------------------------------------

static inline unsigned cdiv(size_t a, size_t b) { return (unsigned)((a + b - 1) / b); }

static void launch_gemm(hipStream_t s, int ta, int tb,
                        const float* A, const float* B, float* C,
                        const float* bias, int bias_mode, const float* resid,
                        int accumulate, int act, float alpha,
                        int M, int N, int K, int lda, int ldb, int ldc) {
  dim3 g(cdiv(N, BN), cdiv(M, BM));
  if (!ta && !tb)
    gemm_wmma_kernel<false, false><<<g, 256, 0, s>>>(A, B, C, bias, resid, M, N, K, lda, ldb, ldc, bias_mode, accumulate, act, alpha);
  else if (ta)
    gemm_wmma_kernel<true, false><<<g, 256, 0, s>>>(A, B, C, bias, resid, M, N, K, lda, ldb, ldc, bias_mode, accumulate, act, alpha);
  else
    gemm_wmma_kernel<false, true><<<g, 256, 0, s>>>(A, B, C, bias, resid, M, N, K, lda, ldb, ldc, bias_mode, accumulate, act, alpha);
}

extern "C" void kernel_launch(void* const* d_in, const int* in_sizes, int n_in,
                              void* d_out, int out_size, void* d_ws, size_t ws_size,
                              hipStream_t stream) {
  (void)in_sizes; (void)n_in; (void)out_size;
  const int L0 = 131072;
  const int fl[7] = {768, 768, 896, 1024, 1152, 1280, 1536};

  // -------- input walker (setup_inputs dict order, depth-first) --------
  int pi = 0;
  const int* ids = (const int*)d_in[pi++];
  auto F = [&]() { return (const float*)d_in[pi++]; };

  const float *stem_w = F(), *stem_b = F();
  const float *scb_g = F(), *scb_b = F(), *scb_rm = F(), *scb_rv = F();
  const float *scb_w = F(), *scb_bb = F();
  const float *stem_pool = F();

  struct Tw { const float *g1,*b1,*rm1,*rv1,*w1,*bb1,*g2,*b2,*rm2,*rv2,*w2,*bb2,*pool; } tw[6];
  for (int s = 0; s < 6; ++s) {
    tw[s].g1 = F(); tw[s].b1 = F(); tw[s].rm1 = F(); tw[s].rv1 = F();
    tw[s].w1 = F(); tw[s].bb1 = F();
    tw[s].g2 = F(); tw[s].b2 = F(); tw[s].rm2 = F(); tw[s].rv2 = F();
    tw[s].w2 = F(); tw[s].bb2 = F();
    tw[s].pool = F();
  }
  struct TL { const float *ln1g,*ln1b,*wq,*wk,*wv,*wo,*bo,*wrk,*rcb,*rpb,*ln2g,*ln2b,*w1,*b1,*w2,*b2; } tl[11];
  for (int l = 0; l < 11; ++l) {
    tl[l].ln1g = F(); tl[l].ln1b = F();
    tl[l].wq = F(); tl[l].wk = F(); tl[l].wv = F(); tl[l].wo = F(); tl[l].bo = F();
    tl[l].wrk = F(); tl[l].rcb = F(); tl[l].rpb = F();
    tl[l].ln2g = F(); tl[l].ln2b = F();
    tl[l].w1 = F(); tl[l].b1 = F(); tl[l].w2 = F(); tl[l].b2 = F();
  }
  const float *fin_g = F(), *fin_b = F(), *fin_rm = F(), *fin_rv = F();
  const float *fin_w = F(), *fin_bb = F();
  const float *head_w = F(), *head_b = F();

  // -------- workspace carve --------
  char* wsb = (char*)d_ws;
  size_t used = 0;
  auto carve = [&](size_t elems) -> float* {
    float* p = (float*)(wsb + used);
    used += ((elems * 4 + 255) / 256) * 256;
    return p;
  };
  const size_t BIGE = (size_t)768 * 131080;   // covers all channel-major stages (+pad)
  float* big0 = carve(BIGE);
  float* big1 = carve(BIGE);
  float* big2 = carve(BIGE);
  float* wpk  = carve((size_t)1536 * 1280 * 5);
  float* prob = carve((size_t)2047 * 32);
  unsigned int* maxb = (unsigned int*)carve(64);
  float* emb  = carve((size_t)2047 * 192);
  float* relk = carve((size_t)2047 * 512);
  float* Xtok = carve((size_t)1024 * 1536);
  float* xln  = carve((size_t)1024 * 1536);
  float* qb   = carve((size_t)1024 * 512);
  float* kb   = carve((size_t)1024 * 512);
  float* vb   = carve((size_t)1024 * 1536);
  float* qc   = carve((size_t)1024 * 512);
  float* qp   = carve((size_t)1024 * 512);
  float* cont = carve((size_t)8 * 1024 * 1024);
  float* relb = carve((size_t)8 * 1024 * 2047);
  float* ctx  = carve((size_t)1024 * 1536);
  float* ffn1 = carve((size_t)1024 * 3072);
  if (used > ws_size) return;  // deterministic no-op if scratch is too small

  int L = L0;

  // -------- stem: conv15 -> (+ conv_block k1) -> attention pool --------
  {
    size_t tot = (size_t)768 * L;
    stem_conv_kernel<<<cdiv(tot, 256), 256, 0, stream>>>(ids, stem_w, stem_b, big0, L);
    bn_gelu_pad_kernel<<<cdiv(tot, 256), 256, 0, stream>>>(big0, scb_g, scb_b, scb_rm, scb_rv, big2, 768, L, 0);
    // big1 = big0 + scb_w @ big2 + bias
    launch_gemm(stream, 0, 0, scb_w, big2, big1, scb_bb, 2, big0, 0, ACT_NONE, 1.f,
                768, L, 768, 768, L, L);
    // pool logits = pool^T @ big1
    launch_gemm(stream, 1, 0, stem_pool, big1, big2, nullptr, 0, nullptr, 0, ACT_NONE, 1.f,
                768, L, 768, 768, L, L);
    attn_pool_kernel<<<cdiv((size_t)768 * (L / 2), 256), 256, 0, stream>>>(big1, big2, big0, 768, L);
    L >>= 1;
  }

  // -------- conv tower (6 stages) --------
  for (int s = 0; s < 6; ++s) {
    int ci = fl[s], co = fl[s + 1];
    // conv_block #1: bn+gelu (padded) then conv5 as 5 accumulating GEMMs
    bn_gelu_pad_kernel<<<cdiv((size_t)ci * (L + 4), 256), 256, 0, stream>>>(
        big0, tw[s].g1, tw[s].b1, tw[s].rm1, tw[s].rv1, big1, ci, L, 2);
    repack_conv5_kernel<<<cdiv((size_t)co * ci * 5, 256), 256, 0, stream>>>(tw[s].w1, wpk, co, ci);
    for (int t = 0; t < 5; ++t)
      launch_gemm(stream, 0, 0, wpk + (size_t)t * co * ci, big1 + t, big2,
                  t == 0 ? tw[s].bb1 : nullptr, t == 0 ? 2 : 0, nullptr, t > 0, ACT_NONE, 1.f,
                  co, L, ci, ci, L + 4, L);
    // conv_block #2 (k=1) with residual: big2 += w2 @ gelu(bn(big2)) + b
    bn_gelu_pad_kernel<<<cdiv((size_t)co * L, 256), 256, 0, stream>>>(
        big2, tw[s].g2, tw[s].b2, tw[s].rm2, tw[s].rv2, big1, co, L, 0);
    launch_gemm(stream, 0, 0, tw[s].w2, big1, big2, tw[s].bb2, 2, nullptr, 1, ACT_NONE, 1.f,
                co, L, co, co, L, L);
    // attention pool
    launch_gemm(stream, 1, 0, tw[s].pool, big2, big1, nullptr, 0, nullptr, 0, ACT_NONE, 1.f,
                co, L, co, co, L, L);
    attn_pool_kernel<<<cdiv((size_t)co * (L / 2), 256), 256, 0, stream>>>(big2, big1, big0, co, L);
    L >>= 1;
  }
  // now big0 = [1536][1024], L == 1024

  // -------- positional embedding (n=1024, 192 features) --------
  init_max_kernel<<<1, 32, 0, stream>>>(maxb);
  gamma_prob_kernel<<<cdiv(2047 * 32, 256), 256, 0, stream>>>(prob, maxb);
  pos_emb_kernel<<<cdiv(2047 * 192, 256), 256, 0, stream>>>(prob, maxb, emb);

  // channel-major -> token-major
  transpose_kernel<<<cdiv((size_t)1536 * 1024, 256), 256, 0, stream>>>(big0, Xtok, 1536, 1024);

  // -------- transformer (11 layers) --------
  for (int l = 0; l < 11; ++l) {
    const TL& P = tl[l];
    layernorm_kernel<<<1024, 256, 0, stream>>>(Xtok, P.ln1g, P.ln1b, xln, 1536);
    launch_gemm(stream, 0, 0, xln, P.wq, qb, nullptr, 0, nullptr, 0, ACT_NONE, 0.125f,
                1024, 512, 1536, 1536, 512, 512);
    launch_gemm(stream, 0, 0, xln, P.wk, kb, nullptr, 0, nullptr, 0, ACT_NONE, 1.f,
                1024, 512, 1536, 1536, 512, 512);
    launch_gemm(stream, 0, 0, xln, P.wv, vb, nullptr, 0, nullptr, 0, ACT_NONE, 1.f,
                1024, 1536, 1536, 1536, 1536, 1536);
    add_bcast_row_kernel<<<cdiv((size_t)1024 * 512, 256), 256, 0, stream>>>(qb, P.rcb, qc, 1024, 512);
    add_bcast_row_kernel<<<cdiv((size_t)1024 * 512, 256), 256, 0, stream>>>(qb, P.rpb, qp, 1024, 512);
    launch_gemm(stream, 0, 0, emb, P.wrk, relk, nullptr, 0, nullptr, 0, ACT_NONE, 1.f,
                2047, 512, 192, 192, 512, 512);
    for (int h = 0; h < 8; ++h) {
      // content[h] = (q+rcb)_h @ k_h^T
      launch_gemm(stream, 0, 1, qc + h * 64, kb + h * 64, cont + (size_t)h * 1024 * 1024,
                  nullptr, 0, nullptr, 0, ACT_NONE, 1.f, 1024, 1024, 64, 512, 512, 1024);
      // rel[h] = (q+rpb)_h @ relk_h^T  (N = 2047)
      launch_gemm(stream, 0, 1, qp + h * 64, relk + h * 64, relb + (size_t)h * 1024 * 2047,
                  nullptr, 0, nullptr, 0, ACT_NONE, 1.f, 1024, 2047, 64, 512, 512, 2047);
    }
    attn_softmax_kernel<<<8 * 1024, 256, 0, stream>>>(cont, relb);
    for (int h = 0; h < 8; ++h)
      launch_gemm(stream, 0, 0, cont + (size_t)h * 1024 * 1024, vb + h * 192, ctx + h * 192,
                  nullptr, 0, nullptr, 0, ACT_NONE, 1.f, 1024, 192, 1024, 1024, 1536, 1536);
    // X += ctx @ wo + bo
    launch_gemm(stream, 0, 0, ctx, P.wo, Xtok, P.bo, 1, nullptr, 1, ACT_NONE, 1.f,
                1024, 1536, 1536, 1536, 1536, 1536);
    // FFN
    layernorm_kernel<<<1024, 256, 0, stream>>>(Xtok, P.ln2g, P.ln2b, xln, 1536);
    launch_gemm(stream, 0, 0, xln, P.w1, ffn1, P.b1, 1, nullptr, 0, ACT_RELU, 1.f,
                1024, 3072, 1536, 1536, 3072, 3072);
    launch_gemm(stream, 0, 0, ffn1, P.w2, Xtok, P.b2, 1, nullptr, 1, ACT_NONE, 1.f,
                1024, 1536, 3072, 3072, 1536, 1536);
  }

  // -------- trim + final conv block + head --------
  // trim rows [64, 960) -> channel-major [1536][896]
  transpose_kernel<<<cdiv((size_t)896 * 1536, 256), 256, 0, stream>>>(Xtok + (size_t)64 * 1536, big0, 896, 1536);
  bn_gelu_pad_kernel<<<cdiv((size_t)1536 * 896, 256), 256, 0, stream>>>(
      big0, fin_g, fin_b, fin_rm, fin_rv, big1, 1536, 896, 0);
  launch_gemm(stream, 0, 0, fin_w, big1, big2, fin_bb, 2, nullptr, 0, ACT_NONE, 1.f,
              3072, 896, 1536, 1536, 896, 896);
  // -> token-major [896][3072], gelu, head GEMM with fused softplus
  transpose_kernel<<<cdiv((size_t)3072 * 896, 256), 256, 0, stream>>>(big2, big0, 3072, 896);
  gelu_kernel<<<cdiv((size_t)896 * 3072, 256), 256, 0, stream>>>(big0, (size_t)896 * 3072);
  launch_gemm(stream, 0, 0, big0, head_w, (float*)d_out, head_b, 1, nullptr, 0, ACT_SOFTPLUS, 1.f,
              896, 5313, 3072, 3072, 5313, 5313);
}
